// MambaTemporalModule_59021440582437
// MI455X (gfx1250) — compile-verified, matrix operand order checked by reading, not run on hardware
//
#include <hip/hip_runtime.h>
#include <hip/hip_bf16.h>

typedef __attribute__((ext_vector_type(16))) __bf16 v16bf;
typedef __attribute__((ext_vector_type(8)))  float  v8f;

// ---------- helpers ----------
__device__ __forceinline__ int kmap(int e, int lane) {
    // 16-bit WMMA A/B operand K mapping (ISA 7.12.2):
    // element e<8 -> K=e, e>=8 -> K=e+8 ; hi half-wave adds 8
    int k = (e & 8) ? (e + 8) : e;
    return k + ((lane & 16) ? 8 : 0);
}
__device__ __forceinline__ float sigmoidf_(float x) { return 1.0f / (1.0f + __expf(-x)); }
__device__ __forceinline__ float siluf_(float x)    { return x * sigmoidf_(x); }
__device__ __forceinline__ float softplusf_(float x){ return (x > 20.0f) ? x : __logf(1.0f + __expf(x)); }

// ---------- weight conversion kernels ----------
__global__ void k_cvt_bf16(const float* __restrict__ src, __bf16* __restrict__ dst, int n) {
    int i = blockIdx.x * blockDim.x + threadIdx.x;
    if (i < n) dst[i] = (__bf16)src[i];
}
__global__ void k_pad_bf16(const float* __restrict__ src, __bf16* __restrict__ dst,
                           int rows, int cols, int prows) {
    int i = blockIdx.x * blockDim.x + threadIdx.x;
    int n = prows * cols;
    if (i < n) {
        int r = i / cols, c = i % cols;
        dst[i] = (r < rows) ? (__bf16)src[r * cols + c] : (__bf16)0.0f;
    }
}

// ---------- Mamba block parameters ----------
struct MP {
    const float*  x;       // (B=4, 16, S, d)
    const __bf16* w_in;    // (2di, d)
    const float*  conv_w;  // (di, 4)
    const float*  conv_b;  // (di)
    const __bf16* w_xp;    // (48, di)  zero-padded past dtr+32 rows
    const float*  dt_w;    // (di, dtr)
    const float*  dt_b;    // (di)
    const float*  A_log;   // (di, 16)
    const float*  Dp;      // (di)
    const __bf16* w_out;   // (d, di)
    int d, di, dtr;
};

// LDS arena layout (bytes). Peak live set (node scan): sXZ+sXDBC+sScr = 43KB.
#define OFF_XZ    0        // float, 16 x 2di   (node: 32KB; dead after scan)
#define OFF_CAT   16384    // float, 16 x 192   (12KB; written in out-GEMM, after sXZ dies;
                           //                    log's sXZ only spans [0,16KB))
#define OFF_XDBC  32768    // float, 16 x 48    (3KB)
#define OFF_SCR   35840    // bf16,  8KB shared by sX -> sXCb -> sYb (disjoint lifetimes)
#define ARENA_SZ  44032

// ---------- one full Mamba block for one sequence (L=16) ----------
// TO_LDS: store y (16 x d fp32) into ldsOut (stride ldsStride)  [fused node/log]
// else  : store directly to gOut[((b*16+t)*S+sn)*d + n]         [trace]
template<bool TO_LDS>
__device__ __forceinline__ void run_mamba(const MP p, int S, int b, int sn,
                                          char* arena, float* ldsOut, int ldsStride,
                                          float* gOut)
{
    float*  sXZ   = (float*) (arena + OFF_XZ);
    float*  sXDBC = (float*) (arena + OFF_XDBC);
    __bf16* sScr  = (__bf16*)(arena + OFF_SCR);   // sX, then sXCb, then sYb

    const int tid    = threadIdx.x;
    const int lane   = tid & 31;
    const int wave   = tid >> 5;
    const int nwaves = blockDim.x >> 5;
    const int d = p.d, di = p.di, dtr = p.dtr;
    const int di2    = 2 * di;
    const int m      = lane & 15;
    const int mbase  = (lane & 16) ? 8 : 0;

    // ---- 1. load X (16 x d) -> LDS bf16 (lives in sScr) ----
    for (int i = tid; i < 16 * d; i += blockDim.x) {
        int t = i / d, k = i % d;
        float v = p.x[(((size_t)b * 16 + t) * S + sn) * d + k];
        sScr[t * d + k] = (__bf16)v;
    }
    __syncthreads();

    // ---- 2. in-proj GEMM: xz(16 x 2di) = X(16 x d) @ w_in^T ----
    for (int tile = wave; tile < (di2 >> 4); tile += nwaves) {
        v8f acc = {0.f,0.f,0.f,0.f,0.f,0.f,0.f,0.f};
        int n = tile * 16 + m;
        for (int k0 = 0; k0 < d; k0 += 32) {
            v16bf fa, fb;
#pragma unroll
            for (int e = 0; e < 16; ++e) {
                int k = k0 + kmap(e, lane);
                fa[e] = sScr[m * d + k];
                fb[e] = p.w_in[(size_t)n * d + k];
            }
            acc = __builtin_amdgcn_wmma_f32_16x16x32_bf16(false, fa, false, fb,
                                                          (short)0, acc, false, false);
        }
#pragma unroll
        for (int v = 0; v < 8; ++v)
            sXZ[(v + mbase) * di2 + n] = acc[v];
    }
    __syncthreads();

    // ---- 3. causal conv (d_conv=4) + SiLU, one channel per thread ----
    if (tid < di) {
        const int c = tid;
        float w0 = p.conv_w[c*4+0], w1 = p.conv_w[c*4+1],
              w2 = p.conv_w[c*4+2], w3 = p.conv_w[c*4+3];
        float cb = p.conv_b[c];
        float xin[16];
#pragma unroll
        for (int t = 0; t < 16; ++t) xin[t] = sXZ[t * di2 + c];
#pragma unroll
        for (int t = 0; t < 16; ++t) {
            float a = cb + xin[t] * w3;
            if (t >= 1) a += xin[t-1] * w2;
            if (t >= 2) a += xin[t-2] * w1;
            if (t >= 3) a += xin[t-3] * w0;
            float xc = siluf_(a);
            sXZ[t * di2 + c] = xc;            // overwrite xin with xc (fp32 for scan)
            sScr[t * di + c] = (__bf16)xc;    // bf16 copy for xproj GEMM (sX dead now)
        }
    }
    __syncthreads();

    // ---- 4. xproj GEMM: xdbc(16 x 48) = xc(16 x di) @ w_xp^T (padded) ----
    if (wave < 3) {
        v8f acc = {0.f,0.f,0.f,0.f,0.f,0.f,0.f,0.f};
        int n = wave * 16 + m;
        for (int k0 = 0; k0 < di; k0 += 32) {
            v16bf fa, fb;
#pragma unroll
            for (int e = 0; e < 16; ++e) {
                int k = k0 + kmap(e, lane);
                fa[e] = sScr[m * di + k];
                fb[e] = p.w_xp[(size_t)n * di + k];
            }
            acc = __builtin_amdgcn_wmma_f32_16x16x32_bf16(false, fa, false, fb,
                                                          (short)0, acc, false, false);
        }
#pragma unroll
        for (int v = 0; v < 8; ++v)
            sXDBC[(v + mbase) * 48 + n] = acc[v];
    }
    __syncthreads();

    // ---- 5. selective scan, one channel per thread; h-state in registers ----
    if (tid < di) {
        const int c = tid;
        float a_s[16];
#pragma unroll
        for (int s = 0; s < 16; ++s) a_s[s] = -__expf(p.A_log[c * 16 + s]);
        float dtw[8];
        for (int j = 0; j < dtr; ++j) dtw[j] = p.dt_w[c * dtr + j];
        const float dtb = p.dt_b[c];
        const float Dc  = p.Dp[c];
        float h[16];
#pragma unroll
        for (int s = 0; s < 16; ++s) h[s] = 0.f;
        for (int t = 0; t < 16; ++t) {
            float lin = dtb;
            for (int j = 0; j < dtr; ++j) lin += sXDBC[t * 48 + j] * dtw[j];
            float dt = softplusf_(lin);
            float u  = sXZ[t * di2 + c];
            float du = dt * u;
            float acc = 0.f;
#pragma unroll
            for (int s = 0; s < 16; ++s) {
                float Bv = sXDBC[t * 48 + dtr + s];
                float Cv = sXDBC[t * 48 + dtr + 16 + s];
                h[s] = __expf(dt * a_s[s]) * h[s] + du * Bv;
                acc += h[s] * Cv;
            }
            float y = acc + u * Dc;
            float z = sXZ[t * di2 + di + c];
            y *= siluf_(z);
            sScr[t * di + c] = (__bf16)y;     // sYb (sXCb dead after xproj)
        }
    }
    __syncthreads();

    // ---- 6. out GEMM: (16 x d) = Y(16 x di) @ w_out^T ----
    for (int tile = wave; tile < (d >> 4); tile += nwaves) {
        v8f acc = {0.f,0.f,0.f,0.f,0.f,0.f,0.f,0.f};
        int n = tile * 16 + m;
        for (int k0 = 0; k0 < di; k0 += 32) {
            v16bf fa, fb;
#pragma unroll
            for (int e = 0; e < 16; ++e) {
                int k = k0 + kmap(e, lane);
                fa[e] = sScr[m * di + k];
                fb[e] = p.w_out[(size_t)n * di + k];
            }
            acc = __builtin_amdgcn_wmma_f32_16x16x32_bf16(false, fa, false, fb,
                                                          (short)0, acc, false, false);
        }
        if (TO_LDS) {
#pragma unroll
            for (int v = 0; v < 8; ++v)
                ldsOut[(v + mbase) * ldsStride + n] = acc[v];   // sXZ already dead
        } else {
#pragma unroll
            for (int v = 0; v < 8; ++v) {
                int t = v + mbase;
                gOut[(((size_t)b * 16 + t) * S + sn) * d + n] = acc[v];
            }
        }
    }
}

// ---------- fused node + log + mix: one workgroup per (b, n) sequence ----------
__global__ __launch_bounds__(256) void fused_node_log_mix(
    MP pn, MP pl, const __bf16* __restrict__ wmix, const float* __restrict__ bias,
    float* __restrict__ out_node, float* __restrict__ out_log, int N)
{
    __shared__ __attribute__((aligned(16))) char arena[ARENA_SZ];
    float*  sCat  = (float*) (arena + OFF_CAT);   // 16 x 192 fp32
    __bf16* sCatB = (__bf16*)(arena);             // 16 x 192 bf16 (after log sXZ dies)

    const int tid  = threadIdx.x;
    const int lane = tid & 31;
    const int wave = tid >> 5;
    const int nwaves = blockDim.x >> 5;
    const int seq = blockIdx.x;
    const int b   = seq / N;
    const int nn  = seq % N;

    run_mamba<true>(pn, N, b, nn, arena, sCat,       192, nullptr);  // cols [0,128)
    __syncthreads();
    run_mamba<true>(pl, N, b, nn, arena, sCat + 128, 192, nullptr);  // cols [128,192)
    __syncthreads();

    for (int i = tid; i < 16 * 192; i += blockDim.x)
        sCatB[i] = (__bf16)sCat[i];
    __syncthreads();

    // mix GEMM: mixed = silu(cat @ mix_W^T + b) + cat, scatter to outputs
    const int m     = lane & 15;
    const int mbase = (lane & 16) ? 8 : 0;
    for (int tile = wave; tile < 12; tile += nwaves) {
        v8f acc = {0.f,0.f,0.f,0.f,0.f,0.f,0.f,0.f};
        int n = tile * 16 + m;
        for (int k0 = 0; k0 < 192; k0 += 32) {
            v16bf fa, fb;
#pragma unroll
            for (int e = 0; e < 16; ++e) {
                int k = k0 + kmap(e, lane);
                fa[e] = sCatB[m * 192 + k];
                fb[e] = wmix[(size_t)n * 192 + k];
            }
            acc = __builtin_amdgcn_wmma_f32_16x16x32_bf16(false, fa, false, fb,
                                                          (short)0, acc, false, false);
        }
        float bn = bias[n];
#pragma unroll
        for (int v = 0; v < 8; ++v) {
            int t = v + mbase;
            float val = siluf_(acc[v] + bn) + sCat[t * 192 + n];
            if (n < 128)
                out_node[(((size_t)b * 16 + t) * N + nn) * 128 + n] = val;
            else
                out_log [(((size_t)b * 16 + t) * N + nn) * 64 + (n - 128)] = val;
        }
    }
}

// ---------- standalone trace block ----------
__global__ __launch_bounds__(128) void trace_kernel(MP p, int S, float* __restrict__ gOut)
{
    __shared__ __attribute__((aligned(16))) char arena[ARENA_SZ];
    const int seq = blockIdx.x;
    const int b   = seq / S;
    const int sn  = seq % S;
    run_mamba<false>(p, S, b, sn, arena, nullptr, 0, gOut);
}

// ---------- host side ----------
extern "C" void kernel_launch(void* const* d_in, const int* in_sizes, int n_in,
                              void* d_out, int out_size, void* d_ws, size_t ws_size,
                              hipStream_t stream) {
    (void)in_sizes; (void)n_in; (void)out_size; (void)ws_size;
    const float* x_node       = (const float*)d_in[0];
    const float* x_trace      = (const float*)d_in[1];
    const float* x_log        = (const float*)d_in[2];
    const float* node_in_w    = (const float*)d_in[3];
    const float* node_conv_w  = (const float*)d_in[4];
    const float* node_conv_b  = (const float*)d_in[5];
    const float* node_xproj_w = (const float*)d_in[6];
    const float* node_dt_w    = (const float*)d_in[7];
    const float* node_dt_b    = (const float*)d_in[8];
    const float* node_A_log   = (const float*)d_in[9];
    const float* node_D       = (const float*)d_in[10];
    const float* node_out_w   = (const float*)d_in[11];
    const float* trace_in_w   = (const float*)d_in[12];
    const float* trace_conv_w = (const float*)d_in[13];
    const float* trace_conv_b = (const float*)d_in[14];
    const float* trace_xproj_w= (const float*)d_in[15];
    const float* trace_dt_w   = (const float*)d_in[16];
    const float* trace_dt_b   = (const float*)d_in[17];
    const float* trace_A_log  = (const float*)d_in[18];
    const float* trace_D      = (const float*)d_in[19];
    const float* trace_out_w  = (const float*)d_in[20];
    const float* log_in_w     = (const float*)d_in[21];
    const float* log_conv_w   = (const float*)d_in[22];
    const float* log_conv_b   = (const float*)d_in[23];
    const float* log_xproj_w  = (const float*)d_in[24];
    const float* log_dt_w     = (const float*)d_in[25];
    const float* log_dt_b     = (const float*)d_in[26];
    const float* log_A_log    = (const float*)d_in[27];
    const float* log_D        = (const float*)d_in[28];
    const float* log_out_w    = (const float*)d_in[29];
    const float* mix_W        = (const float*)d_in[30];
    const float* mix_b        = (const float*)d_in[31];

    const int B = 4, N = 2048, E = 4096;

    // workspace: bf16 weights only (no cat buffer anymore)
    char*  ws  = (char*)d_ws;
    size_t off = 0;
    auto alloc_bf = [&](size_t ne) {
        __bf16* p = (__bf16*)(ws + off);
        off += ((ne * 2 + 255) / 256) * 256;
        return p;
    };
    __bf16* w_in_n  = alloc_bf(512 * 128);
    __bf16* w_xp_n  = alloc_bf(48 * 256);
    __bf16* w_out_n = alloc_bf(128 * 256);
    __bf16* w_in_t  = alloc_bf(256 * 64);
    __bf16* w_xp_t  = alloc_bf(48 * 128);
    __bf16* w_out_t = alloc_bf(64 * 128);
    __bf16* w_in_l  = alloc_bf(256 * 64);
    __bf16* w_xp_l  = alloc_bf(48 * 128);
    __bf16* w_out_l = alloc_bf(64 * 128);
    __bf16* w_mix   = alloc_bf(192 * 192);

    auto cvt = [&](const float* s, __bf16* dst, int n) {
        k_cvt_bf16<<<(n + 255) / 256, 256, 0, stream>>>(s, dst, n);
    };
    cvt(node_in_w,  w_in_n,  512 * 128);
    cvt(node_out_w, w_out_n, 128 * 256);
    cvt(trace_in_w, w_in_t,  256 * 64);
    cvt(trace_out_w,w_out_t, 64 * 128);
    cvt(log_in_w,   w_in_l,  256 * 64);
    cvt(log_out_w,  w_out_l, 64 * 128);
    cvt(mix_W,      w_mix,   192 * 192);
    k_pad_bf16<<<(48 * 256 + 255) / 256, 256, 0, stream>>>(node_xproj_w,  w_xp_n, 40, 256, 48);
    k_pad_bf16<<<(48 * 128 + 255) / 256, 256, 0, stream>>>(trace_xproj_w, w_xp_t, 36, 128, 48);
    k_pad_bf16<<<(48 * 128 + 255) / 256, 256, 0, stream>>>(log_xproj_w,   w_xp_l, 36, 128, 48);

    float* out_node_p  = (float*)d_out;
    float* out_trace_p = (float*)d_out + (size_t)B * 16 * N * 128;
    float* out_log_p   = out_trace_p + (size_t)B * 16 * E * 64;

    MP pn = { x_node, w_in_n, node_conv_w, node_conv_b, w_xp_n, node_dt_w, node_dt_b,
              node_A_log, node_D, w_out_n, 128, 256, 8 };
    MP pl = { x_log,  w_in_l, log_conv_w,  log_conv_b,  w_xp_l, log_dt_w,  log_dt_b,
              log_A_log,  log_D,  w_out_l, 64, 128, 4 };
    MP pt = { x_trace, w_in_t, trace_conv_w, trace_conv_b, w_xp_t, trace_dt_w, trace_dt_b,
              trace_A_log, trace_D, w_out_t, 64, 128, 4 };

    // fused node + log mamba blocks + mix (one wg per (b,n) sequence; cat stays in LDS)
    fused_node_log_mix<<<B * N, 256, 0, stream>>>(pn, pl, w_mix, mix_b,
                                                  out_node_p, out_log_p, N);
    // trace block: direct output
    trace_kernel<<<B * E, 128, 0, stream>>>(pt, E, out_trace_p);
}
